// MultiTimeAttention_45148696215769
// MI455X (gfx1250) — compile-verified, matrix operand order checked by reading, not run on hardware
//
#include <hip/hip_runtime.h>
#include <hip/hip_bf16.h>

// Problem constants (from reference)
constexpr int B_  = 8;
constexpr int LQ_ = 1024;
constexpr int LK_ = 1024;
constexpr int D_  = 512;
constexpr int H_  = 8;
constexpr int DK_ = 64;   // == DV
constexpr float MASKV = -1e30f;

typedef __attribute__((ext_vector_type(16))) _Float16 v16h;
typedef __attribute__((ext_vector_type(8)))  _Float16 v8h;
typedef __attribute__((ext_vector_type(4)))  _Float16 v4h;
typedef __attribute__((ext_vector_type(8)))  float    v8f;
typedef __attribute__((ext_vector_type(4)))  float    v4f;

// ---------------------------------------------------------------------------
// Fragment loader for the CDNA5 16-bit WMMA A/B layout (16x32, wave32):
//   lane L holds row/col (L&15); halves 0..7 at K = 8*(L>>4)..+7,
//   halves 8..15 at K = 16 + 8*(L>>4)..+7.  Two 16-byte loads per fragment
//   (stride must be a multiple of 8 halves to keep chunks 16B aligned).
// ---------------------------------------------------------------------------
__device__ __forceinline__ v16h load_frag(const _Float16* p, int stride,
                                          int row, int kbase) {
    v16h f;
    const _Float16* base = p + (size_t)row * stride + kbase;
    *(v8h*)&f        = *(const v8h*)(base);
    *((v8h*)&f + 1)  = *(const v8h*)(base + 16);
    return f;
}

__device__ __forceinline__ float rowmax16(float x) {
    #pragma unroll
    for (int m = 1; m < 16; m <<= 1) x = fmaxf(x, __shfl_xor(x, m, 16));
    return x;
}
__device__ __forceinline__ float rowsum16(float x) {
    #pragma unroll
    for (int m = 1; m < 16; m <<= 1) x += __shfl_xor(x, m, 16);
    return x;
}

// ---------------------------------------------------------------------------
// Kernel 1: fused Q/K/V projection.  blockIdx.z selects {q,k,v}.
// Each block: 64x64 output tile of one (b,h) GEMM [LQ x D] * [D x 64].
// 128 threads = 4 waves; wave w owns a 32x32 accumulator (2x2 WMMA tiles).
// ---------------------------------------------------------------------------
__global__ __launch_bounds__(128) void qkv_proj_kernel(
    const float* __restrict__ q_in, const float* __restrict__ k_in,
    const float* __restrict__ v_in,
    const float* __restrict__ Wq, const float* __restrict__ Wk,
    const float* __restrict__ Wv,
    _Float16* __restrict__ qh, _Float16* __restrict__ kh,
    _Float16* __restrict__ vh)
{
    const int which = blockIdx.z;
    const float* X = which == 0 ? q_in : (which == 1 ? k_in : v_in);
    const float* W = which == 0 ? Wq   : (which == 1 ? Wk   : Wv);
    _Float16*    O = which == 0 ? qh   : (which == 1 ? kh   : vh);

    const int bh = blockIdx.y;
    const int b  = bh / H_;
    const int h  = bh % H_;
    const int mtile = blockIdx.x * 64;

    const int tid   = threadIdx.x;
    const int lane  = tid & 31;
    const int w     = tid >> 5;
    const int row16 = lane & 15;
    const int kb    = (lane >> 4) * 8;
    const int m0    = (w >> 1) * 32;
    const int n0    = (w & 1) * 32;

    __shared__ _Float16 As[64 * 40];  // [row][k], stride 40 halves (80B)
    __shared__ _Float16 Bs[64 * 40];  // [n][k]  (weight tile transposed)

    const float* Xb = X + (size_t)b * LQ_ * D_;
    const float* Wh = W + (size_t)h * D_ * DK_;

    v8f acc[2][2] = {};

    for (int kk = 0; kk < D_; kk += 32) {
        // prefetch next K-step tiles into cache (global_prefetch_b8)
        const int kkn = kk + 32;
        if (kkn < D_) {
            const int pr = tid & 63;               // one line per A row
            __builtin_prefetch(&Xb[(size_t)(mtile + pr) * D_ + kkn], 0, 3);
            const int pk = tid >> 1, ph = (tid & 1) * 32;  // 2 lines per W row
            __builtin_prefetch(&Wh[(size_t)(kkn + (pk & 31)) * DK_ + ph], 0, 3);
        }

        __syncthreads();
        // stage A: 64 rows x 32 k.  512 float4 loads, 4 per thread,
        // packed v4h (8B) LDS stores.
        #pragma unroll
        for (int it = 0; it < 4; it++) {
            const int f  = tid + 128 * it;       // float4 index 0..511
            const int r  = f >> 3;               // 8 float4 per row
            const int c4 = (f & 7) * 4;
            v4f x = *(const v4f*)&Xb[(size_t)(mtile + r) * D_ + kk + c4];
            v4h hx;
            #pragma unroll
            for (int u = 0; u < 4; u++) hx[u] = (_Float16)x[u];
            *(v4h*)&As[r * 40 + c4] = hx;
        }
        // stage B transposed: Bs[n][k] = W[kk+k][n].  float4 loads,
        // 4 scattered b16 stores each (transpose scatter).
        #pragma unroll
        for (int it = 0; it < 4; it++) {
            const int f  = tid + 128 * it;       // float4 index 0..511
            const int k  = f >> 4;               // 16 float4 per k-row
            const int n4 = (f & 15) * 4;
            v4f x = *(const v4f*)&Wh[(size_t)(kk + k) * DK_ + n4];
            #pragma unroll
            for (int u = 0; u < 4; u++)
                Bs[(n4 + u) * 40 + k] = (_Float16)x[u];
        }
        __syncthreads();

        v16h afr[2], bfr[2];
        #pragma unroll
        for (int i = 0; i < 2; i++)
            afr[i] = load_frag(As, 40, m0 + 16 * i + row16, kb);
        #pragma unroll
        for (int j = 0; j < 2; j++)
            bfr[j] = load_frag(Bs, 40, n0 + 16 * j + row16, kb);

        #pragma unroll
        for (int i = 0; i < 2; i++)
            #pragma unroll
            for (int j = 0; j < 2; j++)
                acc[i][j] = __builtin_amdgcn_wmma_f32_16x16x32_f16(
                    false, afr[i], false, bfr[j], (short)0, acc[i][j],
                    false, false);
    }

    // C layout: VGPR v -> M = v + 8*(lane>>4) = v + kb, N = lane&15
    _Float16* Ob = O + ((size_t)bh * LQ_ + mtile) * DK_;
    #pragma unroll
    for (int i = 0; i < 2; i++)
        #pragma unroll
        for (int j = 0; j < 2; j++)
            #pragma unroll
            for (int vv = 0; vv < 8; vv++) {
                int r = m0 + 16 * i + vv + kb;
                int c = n0 + 16 * j + row16;
                Ob[(size_t)r * DK_ + c] = (_Float16)acc[i][j][vv];
            }
}

// ---------------------------------------------------------------------------
// Kernel 2: flash attention per (b,h, 64-query block).
// 128 threads = 4 waves; wave w owns 16 queries.  Sweep K in 32-key tiles:
//   S = Q K^T (WMMA), mask, online softmax, O += P V (WMMA via LDS bounce).
// ---------------------------------------------------------------------------
__global__ __launch_bounds__(128) void attn_kernel(
    const _Float16* __restrict__ qh, const _Float16* __restrict__ kh,
    const _Float16* __restrict__ vh, const float* __restrict__ mask,
    float* __restrict__ attn)
{
    const int bh = blockIdx.y;
    const int b  = bh >> 3;
    const int h  = bh & 7;

    const int tid   = threadIdx.x;
    const int lane  = tid & 31;
    const int w     = tid >> 5;
    const int row16 = lane & 15;
    const int kb    = (lane >> 4) * 8;
    const int qbase = blockIdx.x * 64 + w * 16;

    const _Float16* Q = qh + (size_t)bh * LQ_ * DK_;
    const _Float16* K = kh + (size_t)bh * LK_ * DK_;
    const _Float16* V = vh + (size_t)bh * LK_ * DK_;
    const float* Mrow = mask + (size_t)b * LK_;

    __shared__ _Float16 Vt[64 * 40];        // [e][key], transposed V tile
    __shared__ _Float16 Plds[4][16 * 40];   // per-wave P bounce buffer

    // Q fragments (16 queries x 64 e) held in registers for the whole sweep
    const int qrow = qbase + row16;
    v16h aq0 = load_frag(Q,      DK_, qrow, kb);   // e 0..31
    v16h aq1 = load_frag(Q + 32, DK_, qrow, kb);   // e 32..63

    v8f o[4] = {};
    float mrow[8], lrow[8];
    #pragma unroll
    for (int vv = 0; vv < 8; vv++) { mrow[vv] = -1e38f; lrow[vv] = 0.0f; }

    for (int kt = 0; kt < LK_; kt += 32) {
        // prefetch next K/V tiles (one 128B line per key row)
        const int ktn = kt + 32;
        if (ktn < LK_) {
            const int pk = tid & 31;
            __builtin_prefetch(&V[(size_t)(ktn + pk) * DK_], 0, 3);
            __builtin_prefetch(&K[(size_t)(ktn + pk) * DK_], 0, 3);
        }

        __syncthreads();
        // stage V tile transposed: Vt[e][key].  256 v8h (16B) global loads,
        // 2 per thread, scattered b16 stores for the transpose.
        #pragma unroll
        for (int it = 0; it < 2; it++) {
            const int f    = tid + 128 * it;    // v8h index 0..255
            const int kidx = f >> 3;            // 8 v8h per key row
            const int e8   = (f & 7) * 8;
            v8h vv8 = *(const v8h*)&V[(size_t)(kt + kidx) * DK_ + e8];
            #pragma unroll
            for (int u = 0; u < 8; u++)
                Vt[(e8 + u) * 40 + kidx] = vv8[u];
        }
        __syncthreads();

        // ---- scores for two 16-key subtiles ----
        v8f s[2];
        #pragma unroll
        for (int t = 0; t < 2; t++) {
            const int key = kt + 16 * t + row16;
            v16h b0 = load_frag(K,      DK_, key, kb);   // e 0..31
            v16h b1 = load_frag(K + 32, DK_, key, kb);   // e 32..63
            v8f sv = {};
            sv = __builtin_amdgcn_wmma_f32_16x16x32_f16(false, aq0, false, b0,
                                                        (short)0, sv, false, false);
            sv = __builtin_amdgcn_wmma_f32_16x16x32_f16(false, aq1, false, b1,
                                                        (short)0, sv, false, false);
            const float mv   = Mrow[key];
            const float bias = (1.0f - mv) * MASKV;
            #pragma unroll
            for (int vv = 0; vv < 8; vv++)
                sv[vv] = mv * sv[vv] + bias;
            s[t] = sv;
        }

        // ---- online softmax update ----
        float mnew[8], scale[8];
        #pragma unroll
        for (int vv = 0; vv < 8; vv++) {
            float tm = rowmax16(fmaxf(s[0][vv], s[1][vv]));
            mnew[vv]  = fmaxf(mrow[vv], tm);
            scale[vv] = __expf(mrow[vv] - mnew[vv]);
            mrow[vv]  = mnew[vv];
        }
        #pragma unroll
        for (int t = 0; t < 2; t++)
            #pragma unroll
            for (int vv = 0; vv < 8; vv++)
                s[t][vv] = __expf(s[t][vv] - mnew[vv]);
        #pragma unroll
        for (int vv = 0; vv < 8; vv++) {
            float rs = rowsum16(s[0][vv] + s[1][vv]);
            lrow[vv] = lrow[vv] * scale[vv] + rs;
        }
        #pragma unroll
        for (int j = 0; j < 4; j++)
            #pragma unroll
            for (int vv = 0; vv < 8; vv++)
                o[j][vv] = o[j][vv] * scale[vv];

        // ---- bounce P through per-wave LDS: C-layout -> A-layout ----
        _Float16* Pw = &Plds[w][0];
        #pragma unroll
        for (int t = 0; t < 2; t++)
            #pragma unroll
            for (int vv = 0; vv < 8; vv++) {
                int r = vv + kb;            // query row within tile
                int c = 16 * t + row16;     // key column within tile
                Pw[r * 40 + c] = (_Float16)s[t][vv];
            }
        v16h pa = load_frag(Pw, 40, row16, kb);   // A-frag 16q x 32key

        // ---- O += P * V ----
        #pragma unroll
        for (int j = 0; j < 4; j++) {
            v16h vb = load_frag(Vt, 40, 16 * j + row16, kb);
            o[j] = __builtin_amdgcn_wmma_f32_16x16x32_f16(false, pa, false, vb,
                                                          (short)0, o[j],
                                                          false, false);
        }
    }

    // normalize and write to concat layout (B, LQ, H*DV)
    float* Ob = attn + (size_t)b * LQ_ * (H_ * DK_);
    #pragma unroll
    for (int vv = 0; vv < 8; vv++) lrow[vv] = 1.0f / lrow[vv];
    #pragma unroll
    for (int j = 0; j < 4; j++)
        #pragma unroll
        for (int vv = 0; vv < 8; vv++) {
            int r = qbase + vv + kb;
            int c = h * DK_ + 16 * j + row16;
            Ob[(size_t)r * (H_ * DK_) + c] = o[j][vv] * lrow[vv];
        }
}

// ---------------------------------------------------------------------------
// Kernel 3: LayerNorm over the 512-wide concat dim, one block per row.
// 128 threads, float4 per thread (b128 load/store).
// ---------------------------------------------------------------------------
__global__ __launch_bounds__(128) void layernorm_kernel(
    const float* __restrict__ attn, const float* __restrict__ gamma,
    const float* __restrict__ beta, float* __restrict__ out)
{
    const int rowi = blockIdx.x;
    const v4f* x = (const v4f*)(attn + (size_t)rowi * 512);
    v4f a = x[threadIdx.x];

    float s  = a[0] + a[1] + a[2] + a[3];
    float s2 = a[0]*a[0] + a[1]*a[1] + a[2]*a[2] + a[3]*a[3];
    #pragma unroll
    for (int m = 1; m < 32; m <<= 1) {
        s  += __shfl_xor(s,  m, 32);
        s2 += __shfl_xor(s2, m, 32);
    }

    __shared__ float rs[4], rs2[4];
    const int w = threadIdx.x >> 5, lane = threadIdx.x & 31;
    if (lane == 0) { rs[w] = s; rs2[w] = s2; }
    __syncthreads();
    s  = rs[0]  + rs[1]  + rs[2]  + rs[3];
    s2 = rs2[0] + rs2[1] + rs2[2] + rs2[3];

    const float mean = s * (1.0f / 512.0f);
    const float var  = s2 * (1.0f / 512.0f) - mean * mean;
    const float inv  = rsqrtf(var + 1e-14f);
    const float g = gamma[0], be = beta[0];

    v4f yv;
    #pragma unroll
    for (int u = 0; u < 4; u++) yv[u] = (a[u] - mean) * inv * g + be;
    ((v4f*)(out + (size_t)rowi * 512))[threadIdx.x] = yv;
}

// ---------------------------------------------------------------------------
// Launch
// ---------------------------------------------------------------------------
extern "C" void kernel_launch(void* const* d_in, const int* in_sizes, int n_in,
                              void* d_out, int out_size, void* d_ws, size_t ws_size,
                              hipStream_t stream) {
    const float* query = (const float*)d_in[0];
    const float* key_t = (const float*)d_in[1];
    const float* value = (const float*)d_in[2];
    const float* mask  = (const float*)d_in[3];
    const float* Wq    = (const float*)d_in[4];
    const float* Wk    = (const float*)d_in[5];
    const float* Wv    = (const float*)d_in[6];
    const float* gamma = (const float*)d_in[7];
    const float* beta  = (const float*)d_in[8];

    // Workspace layout: qh | kh | vh (f16, B*H*LQ*64 each) | attn (f32, B*LQ*512)
    const size_t QKV_BYTES = (size_t)B_ * H_ * LQ_ * DK_ * sizeof(_Float16); // 8 MiB
    char* ws = (char*)d_ws;
    _Float16* qh   = (_Float16*)(ws);
    _Float16* kh   = (_Float16*)(ws + QKV_BYTES);
    _Float16* vh   = (_Float16*)(ws + 2 * QKV_BYTES);
    float*    attn = (float*)   (ws + 3 * QKV_BYTES);

    qkv_proj_kernel<<<dim3(LQ_ / 64, B_ * H_, 3), 128, 0, stream>>>(
        query, key_t, value, Wq, Wk, Wv, qh, kh, vh);

    attn_kernel<<<dim3(LQ_ / 64, B_ * H_), 128, 0, stream>>>(
        qh, kh, vh, mask, attn);

    layernorm_kernel<<<dim3(B_ * LQ_), 128, 0, stream>>>(
        attn, gamma, beta, (float*)d_out);
}
